// HMRInputEncoder_22668837388639
// MI455X (gfx1250) — matrix operand order repaired and checked by reading, not compiled, not run on hardware
//
#include <hip/hip_runtime.h>
#include <cstdint>
#include <cstddef>

typedef __attribute__((ext_vector_type(16))) _Float16 v16h;
typedef __attribute__((ext_vector_type(8)))  _Float16 h8;
typedef __attribute__((ext_vector_type(8)))  float    v8f;

#define HDIM 128
#define NGDF 16
#define KNN  16
#define BB   2
#define NVV  16384
#define NGG  2048
#define BNEPS 1e-5f

static constexpr int  NVB = BB * NVV;                 // 32768 vertices
static constexpr int  NGB = BB * NGG;                 // 4096 graph nodes
static constexpr long long EE = (long long)NVB * KNN; // 524288 edges
static constexpr int  MSGC = HDIM + 2 * NGDF;         // 160 (= 5*32)

// ------------------------------------------------- pad-convert f32 -> f16 [rows, Cpad]
__global__ void k_convpad(const float* __restrict__ x, _Float16* __restrict__ y,
                          int rows, int Cin, int Cpad) {
  long long i = blockIdx.x * (long long)blockDim.x + threadIdx.x;
  if (i >= (long long)rows * Cpad) return;
  int r = (int)(i / Cpad), c = (int)(i % Cpad);
  y[i] = (c < Cin) ? (_Float16)x[(size_t)r * Cin + c] : (_Float16)0.0f;
}

// ------------------------------------------------- pack W[Kd,N] f32 -> WMMA-B layout f16
// Bp element order: e(16) fastest, lane(32), kc(Kpad/32), nt(N/16).
// lane = half*16+lr ; element e of lane's v16h = W[kc*32 + half*16 + e, nt*16 + lr]
__global__ void k_wpack(const float* __restrict__ W, _Float16* __restrict__ Bp,
                        int Kd, int N, int Kpad) {
  long long i = blockIdx.x * (long long)blockDim.x + threadIdx.x;
  long long total = (long long)(N / 16) * (Kpad / 32) * 512;
  if (i >= total) return;
  int e    = (int)(i & 15);
  int lane = (int)((i >> 4) & 31);
  int kc   = (int)((i >> 9) % (Kpad >> 5));
  int nt   = (int)(i / ((long long)(Kpad >> 5) << 9));
  int half = lane >> 4, lr = lane & 15;
  int k = kc * 32 + half * 16 + e;
  int n = nt * 16 + lr;
  Bp[i] = (k < Kd) ? (_Float16)W[(size_t)k * N + n] : (_Float16)0.0f;
}

__global__ void k_zero_stats(float* __restrict__ s, float* __restrict__ s2) {
  int i = threadIdx.x; s[i] = 0.f; s2[i] = 0.f;
}

// -------------------------------------------------- WMMA GEMM  C = A(f16,[M,Kpad]) * Bpacked
// block (32,2): 2 waves; each wave computes 16 rows x 64 cols (4 n-tiles, 4 accumulators).
// grid = (M/16, N/128). No guards anywhere: EXEC stays all-ones around every v_wmma.
__global__ void k_gemm_wmma(const _Float16* __restrict__ A, const _Float16* __restrict__ Bp,
                            _Float16* __restrict__ C, int M, int N, int Kpad) {
  const int lane = threadIdx.x;
  const int half = lane >> 4;
  const int lr   = lane & 15;
  const int m0   = blockIdx.x * 16;
  const int nw0  = (blockIdx.y * blockDim.y + threadIdx.y) * 4; // first of 4 n-tiles
  const int KC   = Kpad >> 5;

  v8f acc0 = {}, acc1 = {}, acc2 = {}, acc3 = {};
  const _Float16* arow = A + (size_t)(m0 + lr) * Kpad;
  const v16h* bb = (const v16h*)Bp;
  const size_t b0o = ((size_t)(nw0 + 0) * KC) * 32 + lane;
  const size_t b1o = ((size_t)(nw0 + 1) * KC) * 32 + lane;
  const size_t b2o = ((size_t)(nw0 + 2) * KC) * 32 + lane;
  const size_t b3o = ((size_t)(nw0 + 3) * KC) * 32 + lane;

  for (int kc = 0; kc < KC; ++kc) {
    const int kb = kc << 5;
    v16h a = *(const v16h*)(arow + kb + half * 16);    // K = kb + half*16 + 0..15
    v16h b0 = bb[b0o + (size_t)kc * 32];
    v16h b1 = bb[b1o + (size_t)kc * 32];
    v16h b2 = bb[b2o + (size_t)kc * 32];
    v16h b3 = bb[b3o + (size_t)kc * 32];
    acc0 = __builtin_amdgcn_wmma_f32_16x16x32_f16(false, a, false, b0, (short)0, acc0, false, false);
    acc1 = __builtin_amdgcn_wmma_f32_16x16x32_f16(false, a, false, b1, (short)0, acc1, false, false);
    acc2 = __builtin_amdgcn_wmma_f32_16x16x32_f16(false, a, false, b2, (short)0, acc2, false, false);
    acc3 = __builtin_amdgcn_wmma_f32_16x16x32_f16(false, a, false, b3, (short)0, acc3, false, false);
  }
  // C/D layout: VGPR i -> row m0 + half*8 + i, col = ntile*16 + lr
#pragma unroll
  for (int i = 0; i < 8; ++i) {
    size_t ro = (size_t)(m0 + half * 8 + i) * N + lr;
    C[ro + (size_t)(nw0 + 0) * 16] = (_Float16)acc0[i];
    C[ro + (size_t)(nw0 + 1) * 16] = (_Float16)acc1[i];
    C[ro + (size_t)(nw0 + 2) * 16] = (_Float16)acc2[i];
    C[ro + (size_t)(nw0 + 3) * 16] = (_Float16)acc3[i];
  }
}

// ---------------------------------------------------------- BN batch statistics
__global__ void k_colstats(const _Float16* __restrict__ X, int C, long long Nrows,
                           float* __restrict__ sum, float* __restrict__ sumsq) {
  int c = threadIdx.x;
  float s = 0.f, s2 = 0.f;
  for (long long r = blockIdx.x; r < Nrows; r += gridDim.x) {
    float v = (float)X[r * C + c];
    s += v; s2 += v * v;
  }
  atomicAdd(&sum[c], s);
  atomicAdd(&sumsq[c], s2);
}

__global__ void k_bn_finalize(const float* __restrict__ sum, const float* __restrict__ sumsq,
                              const float* __restrict__ g, const float* __restrict__ be,
                              float Nrows, float* __restrict__ scale, float* __restrict__ shift, int C) {
  int c = threadIdx.x;
  if (c < C) {
    float m  = sum[c] / Nrows;
    float v  = sumsq[c] / Nrows - m * m;
    float sc = g[c] * rsqrtf(v + BNEPS);
    scale[c] = sc;
    shift[c] = be[c] - m * sc;
  }
}

// y = bn(x) [optionally SiLU], f16 out (in-place safe)
__global__ void k_bn_apply(const _Float16* __restrict__ X, _Float16* __restrict__ Y,
                           const float* __restrict__ scale, const float* __restrict__ shift,
                           int C, long long n, int silu) {
  long long i = blockIdx.x * (long long)blockDim.x + threadIdx.x;
  if (i >= n) return;
  int c = (int)(i % C);
  float y = (float)X[i] * scale[c] + shift[c];
  if (silu) y = y / (1.f + expf(-y));
  Y[i] = (_Float16)y;
}

// final layer: bn only, f32 out
__global__ void k_bn_out_f32(const _Float16* __restrict__ X, float* __restrict__ Y,
                             const float* __restrict__ scale, const float* __restrict__ shift,
                             int C, long long n) {
  long long i = blockIdx.x * (long long)blockDim.x + threadIdx.x;
  if (i >= n) return;
  int c = (int)(i % C);
  Y[i] = (float)X[i] * scale[c] + shift[c];
}

__device__ __forceinline__ float d_sigmoid(float x) { return 1.f / (1.f + expf(-x)); }
__device__ __forceinline__ float d_softplus(float x) { return (x > 20.f) ? x : log1pf(expf(x)); }

// chem gate: h = sigmoid(bn(f)) * softplus(bn(c)); write f32 (output) + f16 (for edge gather)
__global__ void k_chem_gate(const _Float16* __restrict__ Y2,
                            const float* __restrict__ scale, const float* __restrict__ shift,
                            float* __restrict__ out32, _Float16* __restrict__ out16) {
  long long i = blockIdx.x * (long long)blockDim.x + threadIdx.x;
  if (i >= (long long)NGB * HDIM) return;
  int r = (int)(i / HDIM), c = (int)(i % HDIM);
  float f  = (float)Y2[(size_t)r * 2 * HDIM + c]        * scale[c]        + shift[c];
  float cc = (float)Y2[(size_t)r * 2 * HDIM + HDIM + c] * scale[HDIM + c] + shift[HDIM + c];
  float v = d_sigmoid(f) * d_softplus(cc);
  out32[i] = v;
  out16[i] = (_Float16)v;
}

// ------------------------------------------------------------------- KNN (top-16 smallest d2)
__global__ void k_knn(const float* __restrict__ verts, const float* __restrict__ nodes,
                      int* __restrict__ knn_idx) {
  int t = blockIdx.x * blockDim.x + threadIdx.x;
  if (t >= NVB) return;
  int b = t / NVV;
  float vx = verts[(size_t)t * 3], vy = verts[(size_t)t * 3 + 1], vz = verts[(size_t)t * 3 + 2];
  float bd[KNN]; int bi[KNN];
#pragma unroll
  for (int j = 0; j < KNN; ++j) { bd[j] = 3.4e38f; bi[j] = 0; }
  float worst = 3.4e38f; int wpos = 0;
  const float* npb = nodes + (size_t)b * NGG * 3;
  for (int g = 0; g < NGG; ++g) {
    float dx = npb[g * 3] - vx, dy = npb[g * 3 + 1] - vy, dz = npb[g * 3 + 2] - vz;
    float d2 = dx * dx + dy * dy + dz * dz;
    if (d2 < worst) {
      bd[wpos] = d2; bi[wpos] = g;
      worst = bd[0]; wpos = 0;
#pragma unroll
      for (int j = 1; j < KNN; ++j) if (bd[j] > worst) { worst = bd[j]; wpos = j; }
    }
  }
#pragma unroll
  for (int j = 0; j < KNN; ++j) knn_idx[(size_t)t * KNN + j] = b * NGG + bi[j];
}

// --------------------------------------------- build msg rows [E, 160] (f16)
__global__ void k_msg(const int* __restrict__ knn_idx, const _Float16* __restrict__ hchem,
                      const float* __restrict__ verts, const float* __restrict__ nodes,
                      const float* __restrict__ vnorm, _Float16* __restrict__ msg) {
  long long e = blockIdx.x;
  int c = threadIdx.x;
  int g = knn_idx[e];
  _Float16* row = msg + e * MSGC;
  if (c < HDIM) { row[c] = hchem[(size_t)g * HDIM + c]; return; }
  int v = (int)(e / KNN);
  float dx = nodes[(size_t)g * 3]     - verts[(size_t)v * 3];
  float dy = nodes[(size_t)g * 3 + 1] - verts[(size_t)v * 3 + 1];
  float dz = nodes[(size_t)g * 3 + 2] - verts[(size_t)v * 3 + 2];
  float dist = sqrtf(dx * dx + dy * dy + dz * dz);
  float inv = 1.f / dist;
  float ang = (dx * vnorm[(size_t)v * 3] + dy * vnorm[(size_t)v * 3 + 1] + dz * vnorm[(size_t)v * 3 + 2]) * inv;
  int j = c - HDIM;
  float val;
  if (j < NGDF) { float mu = (float)j * (8.f / 15.f);                 float t = (dist - mu) / 0.5f;   val = expf(-t * t); }
  else          { float mu = -1.f + (float)(j - NGDF) * (2.f / 15.f); float t = (ang - mu) / 0.125f;  val = expf(-t * t); }
  row[c] = (_Float16)val;
}

// surf gate + segment-sum over the 16 contiguous edges of each vertex;
// also writes geom half of the feat-MLP input concat.
__global__ void k_surf_gate_segsum(const _Float16* __restrict__ Y2,
                                   const float* __restrict__ scale, const float* __restrict__ shift,
                                   const _Float16* __restrict__ geomh, _Float16* __restrict__ featin) {
  long long i = blockIdx.x * (long long)blockDim.x + threadIdx.x;
  if (i >= (long long)NVB * HDIM) return;
  int v = (int)(i / HDIM), c = (int)(i % HDIM);
  float scf = scale[c], shf = shift[c], scc = scale[HDIM + c], shc = shift[HDIM + c];
  float acc = 0.f;
  for (int k = 0; k < KNN; ++k) {
    size_t e = (size_t)v * KNN + k;
    float f  = (float)Y2[e * 2 * HDIM + c]        * scf + shf;
    float cc = (float)Y2[e * 2 * HDIM + HDIM + c] * scc + shc;
    acc += d_sigmoid(f) * d_softplus(cc);
  }
  featin[(size_t)v * 2 * HDIM + c]        = (_Float16)acc;
  featin[(size_t)v * 2 * HDIM + HDIM + c] = geomh[(size_t)v * HDIM + c];
}

// ------------------------------------------------------------------- driver
static inline unsigned gb(long long n, int b) { return (unsigned)((n + b - 1) / b); }

extern "C" void kernel_launch(void* const* d_in, const int* in_sizes, int n_in,
                              void* d_out, int out_size, void* d_ws, size_t ws_size,
                              hipStream_t stream) {
  (void)in_sizes; (void)n_in; (void)out_size; (void)ws_size;
  const float* chem_feats = (const float*)d_in[0];
  const float* geom_feats = (const float*)d_in[1];
  const float* verts      = (const float*)d_in[2];
  const float* node_pos   = (const float*)d_in[3];
  const float* vnormals   = (const float*)d_in[4];
  // per-MLP params: w1,b1,g1,be1,w2,b2,g2,be2  (linear biases cancel under training BN -> unused)
  const float* chem_w1 = (const float*)d_in[5];  const float* chem_g1 = (const float*)d_in[7];
  const float* chem_be1= (const float*)d_in[8];  const float* chem_w2 = (const float*)d_in[9];
  const float* chem_g2 = (const float*)d_in[11]; const float* chem_be2= (const float*)d_in[12];
  const float* geom_w1 = (const float*)d_in[13]; const float* geom_g1 = (const float*)d_in[15];
  const float* geom_be1= (const float*)d_in[16]; const float* geom_w2 = (const float*)d_in[17];
  const float* geom_g2 = (const float*)d_in[19]; const float* geom_be2= (const float*)d_in[20];
  const float* surf_w1 = (const float*)d_in[21]; const float* surf_g1 = (const float*)d_in[23];
  const float* surf_be1= (const float*)d_in[24]; const float* surf_w2 = (const float*)d_in[25];
  const float* surf_g2 = (const float*)d_in[27]; const float* surf_be2= (const float*)d_in[28];
  const float* feat_w1 = (const float*)d_in[29]; const float* feat_g1 = (const float*)d_in[31];
  const float* feat_be1= (const float*)d_in[32]; const float* feat_w2 = (const float*)d_in[33];
  const float* feat_g2 = (const float*)d_in[35]; const float* feat_be2= (const float*)d_in[36];

  float* out_hgeom = (float*)d_out;
  float* out_hchem = out_hgeom + (size_t)NVB * HDIM;

  // workspace carve-up
  uint8_t* base = (uint8_t*)d_ws; size_t off = 0;
  auto alloc = [&](size_t bytes) -> void* { void* p = base + off; off += (bytes + 255) & ~(size_t)255; return p; };
  float*    f_sum   = (float*)alloc(256 * 4);
  float*    f_sumsq = (float*)alloc(256 * 4);
  float*    f_scale = (float*)alloc(256 * 4);
  float*    f_shift = (float*)alloc(256 * 4);
  _Float16* chemA   = (_Float16*)alloc((size_t)NGB * 32 * 2);     // padded 30->32
  _Float16* geomA   = (_Float16*)alloc((size_t)NVB * 32 * 2);     // padded 22->32
  _Float16* chem_y1 = (_Float16*)alloc((size_t)NGB * HDIM * 2);
  _Float16* chem_y2 = (_Float16*)alloc((size_t)NGB * 2 * HDIM * 2);
  _Float16* hchem16 = (_Float16*)alloc((size_t)NGB * HDIM * 2);
  _Float16* geom_y1 = (_Float16*)alloc((size_t)NVB * HDIM * 2);
  _Float16* geom_y2 = (_Float16*)alloc((size_t)NVB * HDIM * 2);
  _Float16* geom_h  = (_Float16*)alloc((size_t)NVB * HDIM * 2);
  int*      knnidx  = (int*)alloc((size_t)EE * 4);
  _Float16* featin  = (_Float16*)alloc((size_t)NVB * 2 * HDIM * 2);
  _Float16* feat_y1 = (_Float16*)alloc((size_t)NVB * HDIM * 2);
  _Float16* feat_y2 = (_Float16*)alloc((size_t)NVB * HDIM * 2);
  // packed weights (WMMA-B layout), Kpad x N f16 each
  _Float16* p_chem_w1 = (_Float16*)alloc((size_t)32  * 128 * 2);
  _Float16* p_chem_w2 = (_Float16*)alloc((size_t)128 * 256 * 2);
  _Float16* p_geom_w1 = (_Float16*)alloc((size_t)32  * 128 * 2);
  _Float16* p_geom_w2 = (_Float16*)alloc((size_t)128 * 128 * 2);
  _Float16* p_surf_w1 = (_Float16*)alloc((size_t)160 * 128 * 2);
  _Float16* p_surf_w2 = (_Float16*)alloc((size_t)128 * 256 * 2);
  _Float16* p_feat_w1 = (_Float16*)alloc((size_t)256 * 128 * 2);
  _Float16* p_feat_w2 = (_Float16*)alloc((size_t)128 * 128 * 2);
  _Float16* msg     = (_Float16*)alloc((size_t)EE * MSGC * 2);      // 160 MB
  _Float16* surf_y1 = (_Float16*)alloc((size_t)EE * HDIM * 2);      // 128 MB
  _Float16* surf_y2 = (_Float16*)alloc((size_t)EE * 2 * HDIM * 2);  // 256 MB

  auto wpack = [&](const float* W, _Float16* Bp, int Kd, int N, int Kpad) {
    long long total = (long long)(N / 16) * (Kpad / 32) * 512;
    k_wpack<<<gb(total, 256), 256, 0, stream>>>(W, Bp, Kd, N, Kpad);
  };
  auto gemm = [&](const _Float16* A, const _Float16* Bp, _Float16* C, int M, int N, int Kpad) {
    dim3 blk(32, 2);                 // 2 waves x (16 rows x 64 cols)
    dim3 grd(M / 16, N / 128);
    k_gemm_wmma<<<grd, blk, 0, stream>>>(A, Bp, C, M, N, Kpad);
  };
  auto bnstats = [&](const _Float16* X, int C, long long Nrows, const float* g, const float* be) {
    k_zero_stats<<<1, 256, 0, stream>>>(f_sum, f_sumsq);
    int blocks = (Nrows > 65536) ? 1024 : 128;
    k_colstats<<<blocks, C, 0, stream>>>(X, C, Nrows, f_sum, f_sumsq);
    k_bn_finalize<<<1, 256, 0, stream>>>(f_sum, f_sumsq, g, be, (float)Nrows, f_scale, f_shift, C);
  };

  // ---- weight packing (tiny; reused by up to 32768 M-tiles each)
  wpack(chem_w1, p_chem_w1, 30,  128, 32);
  wpack(chem_w2, p_chem_w2, 128, 256, 128);
  wpack(geom_w1, p_geom_w1, 22,  128, 32);
  wpack(geom_w2, p_geom_w2, 128, 128, 128);
  wpack(surf_w1, p_surf_w1, 160, 128, 160);
  wpack(surf_w2, p_surf_w2, 128, 256, 128);
  wpack(feat_w1, p_feat_w1, 256, 128, 256);
  wpack(feat_w2, p_feat_w2, 128, 128, 128);

  // ---- input conversion (zero-padded K)
  k_convpad<<<gb((long long)NGB * 32, 256), 256, 0, stream>>>(chem_feats, chemA, NGB, 30, 32);
  k_convpad<<<gb((long long)NVB * 32, 256), 256, 0, stream>>>(geom_feats, geomA, NVB, 22, 32);

  // ---- chem MLP: 30 -> 128 (BN,SiLU) -> 256 (BN) -> gate
  gemm(chemA, p_chem_w1, chem_y1, NGB, HDIM, 32);
  bnstats(chem_y1, HDIM, NGB, chem_g1, chem_be1);
  k_bn_apply<<<gb((long long)NGB * HDIM, 256), 256, 0, stream>>>(chem_y1, chem_y1, f_scale, f_shift, HDIM, (long long)NGB * HDIM, 1);
  gemm(chem_y1, p_chem_w2, chem_y2, NGB, 2 * HDIM, HDIM);
  bnstats(chem_y2, 2 * HDIM, NGB, chem_g2, chem_be2);
  k_chem_gate<<<gb((long long)NGB * HDIM, 256), 256, 0, stream>>>(chem_y2, f_scale, f_shift, out_hchem, hchem16);

  // ---- geom MLP: 22 -> 128 (BN,SiLU) -> 128 (BN)
  gemm(geomA, p_geom_w1, geom_y1, NVB, HDIM, 32);
  bnstats(geom_y1, HDIM, NVB, geom_g1, geom_be1);
  k_bn_apply<<<gb((long long)NVB * HDIM, 256), 256, 0, stream>>>(geom_y1, geom_y1, f_scale, f_shift, HDIM, (long long)NVB * HDIM, 1);
  gemm(geom_y1, p_geom_w2, geom_y2, NVB, HDIM, HDIM);
  bnstats(geom_y2, HDIM, NVB, geom_g2, geom_be2);
  k_bn_apply<<<gb((long long)NVB * HDIM, 256), 256, 0, stream>>>(geom_y2, geom_h, f_scale, f_shift, HDIM, (long long)NVB * HDIM, 0);

  // ---- KNN + edge message build
  k_knn<<<gb(NVB, 256), 256, 0, stream>>>(verts, node_pos, knnidx);
  k_msg<<<dim3((unsigned)EE), MSGC, 0, stream>>>(knnidx, hchem16, verts, node_pos, vnormals, msg);

  // ---- surf MLP on edges: 160 -> 128 (BN,SiLU) -> 256 (BN) -> gate + segment-sum
  gemm(msg, p_surf_w1, surf_y1, (int)EE, HDIM, MSGC);
  bnstats(surf_y1, HDIM, EE, surf_g1, surf_be1);
  k_bn_apply<<<gb(EE * HDIM, 256), 256, 0, stream>>>(surf_y1, surf_y1, f_scale, f_shift, HDIM, EE * HDIM, 1);
  gemm(surf_y1, p_surf_w2, surf_y2, (int)EE, 2 * HDIM, HDIM);
  bnstats(surf_y2, 2 * HDIM, EE, surf_g2, surf_be2);
  k_surf_gate_segsum<<<gb((long long)NVB * HDIM, 256), 256, 0, stream>>>(surf_y2, f_scale, f_shift, geom_h, featin);

  // ---- feat MLP: 256 -> 128 (BN,SiLU) -> 128 (BN) -> f32 output
  gemm(featin, p_feat_w1, feat_y1, NVB, HDIM, 2 * HDIM);
  bnstats(feat_y1, HDIM, NVB, feat_g1, feat_be1);
  k_bn_apply<<<gb((long long)NVB * HDIM, 256), 256, 0, stream>>>(feat_y1, feat_y1, f_scale, f_shift, HDIM, (long long)NVB * HDIM, 1);
  gemm(feat_y1, p_feat_w2, feat_y2, NVB, HDIM, HDIM);
  bnstats(feat_y2, HDIM, NVB, feat_g2, feat_be2);
  k_bn_out_f32<<<gb((long long)NVB * HDIM, 256), 256, 0, stream>>>(feat_y2, out_hgeom, f_scale, f_shift, HDIM, (long long)NVB * HDIM);
}
